// RNNModel_37881611550912
// MI455X (gfx1250) — matrix operand order repaired and checked by reading, not compiled, not run on hardware
//
#include <hip/hip_runtime.h>
#include <hip/hip_bf16.h>

#define B_ 1024
#define T_ 100
#define F_ 300
#define U_ 512
#define C_ 10
#define GW 2048          // 4*U
#define EPS_ 1e-3f

typedef __attribute__((ext_vector_type(16))) __bf16 v16bf;
typedef __attribute__((ext_vector_type(8)))  float  v8f;

// ---------------------------------------------------------------------------
// Weight packing: W (K x 2048 row-major f32) -> bf16 B-fragment order.
// B tile 32x16: lane l holds column n=(l&15); k = kt*32 + (l<16?0:8) + e + (e>=8 ? 8 : 0).
// Buffer index: (((ntG*ktiles + kt)*32 + lane)*16 + e)
// ---------------------------------------------------------------------------
__global__ void pack_weights(const float* __restrict__ src,
                             __hip_bfloat16* __restrict__ dst,
                             int korig, int ktiles) {
    long idx = (long)blockIdx.x * 256 + threadIdx.x;
    long total = (long)ktiles * 32 * 16 * (GW / 16);
    if (idx >= total) return;
    int e    = (int)(idx & 15);
    int lane = (int)((idx >> 4) & 31);
    int kt   = (int)((idx >> 9) % ktiles);
    int ntG  = (int)(idx / ((long)ktiles << 9));
    int n = ntG * 16 + (lane & 15);
    int k = kt * 32 + ((lane < 16) ? 0 : 8) + e + ((e >= 8) ? 8 : 0);
    float v = (k < korig) ? src[(long)k * GW + n] : 0.0f;
    dst[idx] = __float2bfloat16(v);
}

// ---------------------------------------------------------------------------
// Device helpers
// ---------------------------------------------------------------------------
__device__ __forceinline__ float half16_reduce(float v) {
    v += __shfl_xor(v, 1, 32);
    v += __shfl_xor(v, 2, 32);
    v += __shfl_xor(v, 4, 32);
    v += __shfl_xor(v, 8, 32);
    return v;   // sum within each 16-lane half
}
__device__ __forceinline__ float wave_reduce(float v) {
#pragma unroll
    for (int off = 16; off; off >>= 1) v += __shfl_xor(v, off, 32);
    return v;
}
__device__ __forceinline__ float sigm(float x) { return 1.0f / (1.0f + expf(-x)); }

// A-fragment layout (16x32 bf16 tile):
//   lane l: M = l&15 ; k = kt*32 + (l<16?0:8) + e + (e>=8 ? 8 : 0)
// Inverse: element index of (m, k) in a packed A buffer:
__device__ __forceinline__ int apack_idx(int m, int k) {
    int kt = k >> 5, r = k & 31;
    int half_ = (r >> 3) & 1;            // 0..7->0, 8..15->1, 16..23->0, 24..31->1
    int e = (r & 7) | ((r >> 1) & 8);    // +8 for r>=16
    return (((kt << 5) + (m & 15) + (half_ << 4)) << 4) + e;
}

struct LayerParams {
    const __hip_bfloat16* Wpk;
    const __hip_bfloat16* Rpk;
    const float *bias, *kg, *kb, *rg, *rb, *sg, *sb;
    int ktk_x;     // K tiles for the input (x) GEMM
};

// ---------------------------------------------------------------------------
// One LSTM layer, one timestep, for this block's 16 rows.
// stat rows: 0 kSum 1 kSq 2 rSum 3 rSq 4 kMean 5 kRstd 6 rMean 7 rRstd
//            8 cSum 9 cSq 10 cMean 11 cRstd
// ---------------------------------------------------------------------------
__device__ __forceinline__ void layer_step(
    const LayerParams& P,
    const v16bf* apk,                    // packed A frags, K path (LDS)
    const v16bf* hpk,                    // packed A frags, R path (LDS)
    float (*hs)[U_],                     // recurrent state h f32 (LDS)
    float* creg,                         // [16] cell state (registers, u = tid)
    __bf16* hpk_out,                     // write masked state, packed (LDS)
    __bf16* rawpk_out,                   // layer0: raw hn packed ; else null
    float* outreg,                       // layer1: [16] masked output ; else null
    float (*zs)[GW], float (*stat)[16],
    const int* mask, int b0, int t) {

    const int tid   = threadIdx.x;
    const int lane  = tid & 31;
    const int w     = tid >> 5;
    const int wbase = w * 128;
    const int half_ = (lane >> 4);       // 0 or 1
    const int nlane = lane & 15;

    if (tid < 192) stat[tid >> 4][tid & 15] = 0.0f;
    __syncthreads();

    v8f zero8 = {};
    v8f acc[8];
#pragma unroll
    for (int i = 0; i < 8; ++i) acc[i] = zero8;

    // ---- K path: Ain @ W -------------------------------------------------
    const v16bf* wp = reinterpret_cast<const v16bf*>(P.Wpk);
    const int ktk = P.ktk_x;
    for (int kt = 0; kt < ktk; ++kt) {
        v16bf a = apk[(kt << 5) + lane];            // one 32B LDS load
        v16bf b[8];
#pragma unroll
        for (int nt = 0; nt < 8; ++nt)              // batched global loads
            b[nt] = wp[((long)(w * 8 + nt) * ktk + kt) * 32 + lane];
#pragma unroll
        for (int nt = 0; nt < 8; ++nt)              // back-to-back WMMA
            acc[nt] = __builtin_amdgcn_wmma_f32_16x16x32_bf16(
                false, a, false, b[nt], (short)0, acc[nt], false, false);
    }
    // row stats + stash raw xW into zs
#pragma unroll
    for (int v = 0; v < 8; ++v) {
        int row = v + half_ * 8;
        float s = 0.f, q = 0.f;
#pragma unroll
        for (int nt = 0; nt < 8; ++nt) {
            float x = acc[nt][v];
            s += x; q += x * x;
            zs[row][wbase + nt * 16 + nlane] = x;
        }
        s = half16_reduce(s); q = half16_reduce(q);
        if (nlane == 0) { atomicAdd(&stat[0][row], s); atomicAdd(&stat[1][row], q); }
    }

    // ---- R path: h @ R ---------------------------------------------------
#pragma unroll
    for (int i = 0; i < 8; ++i) acc[i] = zero8;
    const v16bf* rp = reinterpret_cast<const v16bf*>(P.Rpk);
    for (int kt = 0; kt < 16; ++kt) {
        v16bf a = hpk[(kt << 5) + lane];
        v16bf b[8];
#pragma unroll
        for (int nt = 0; nt < 8; ++nt)
            b[nt] = rp[((long)(w * 8 + nt) * 16 + kt) * 32 + lane];
#pragma unroll
        for (int nt = 0; nt < 8; ++nt)
            acc[nt] = __builtin_amdgcn_wmma_f32_16x16x32_bf16(
                false, a, false, b[nt], (short)0, acc[nt], false, false);
    }
#pragma unroll
    for (int v = 0; v < 8; ++v) {
        int row = v + half_ * 8;
        float s = 0.f, q = 0.f;
#pragma unroll
        for (int nt = 0; nt < 8; ++nt) { float x = acc[nt][v]; s += x; q += x * x; }
        s = half16_reduce(s); q = half16_reduce(q);
        if (nlane == 0) { atomicAdd(&stat[2][row], s); atomicAdd(&stat[3][row], q); }
    }
    __syncthreads();

    if (tid < 16) {
        float mk = stat[0][tid] * (1.0f / GW);
        float vk = stat[1][tid] * (1.0f / GW) - mk * mk;
        stat[4][tid] = mk; stat[5][tid] = rsqrtf(vk + EPS_);
        float mr = stat[2][tid] * (1.0f / GW);
        float vr = stat[3][tid] * (1.0f / GW) - mr * mr;
        stat[6][tid] = mr; stat[7][tid] = rsqrtf(vr + EPS_);
    }
    __syncthreads();

    // ---- combine: z = LN(xW)*kg+kb + LN(hR)*rg+rb + bias ----------------
#pragma unroll
    for (int v = 0; v < 8; ++v) {
        int row = v + half_ * 8;
        float km = stat[4][row], ks = stat[5][row];
        float rm = stat[6][row], rs = stat[7][row];
#pragma unroll
        for (int nt = 0; nt < 8; ++nt) {
            int n = wbase + nt * 16 + nlane;
            float zk = zs[row][n];
            float z = (zk - km) * ks * P.kg[n] + P.kb[n]
                    + (acc[nt][v] - rm) * rs * P.rg[n] + P.rb[n] + P.bias[n];
            zs[row][n] = z;
        }
    }
    __syncthreads();

    // ---- gates: thread tid owns u = tid for all 16 rows ------------------
    const int u = tid;
    float cn[16];
#pragma unroll
    for (int row = 0; row < 16; ++row) {
        float zi = zs[row][u];
        float zf = zs[row][u + 512];
        float zg = zs[row][u + 1024];
        float cv = sigm(zf) * creg[row] + sigm(zi) * tanhf(zg);
        cn[row] = cv;
        float s = wave_reduce(cv);
        float q = wave_reduce(cv * cv);
        if (lane == 0) { atomicAdd(&stat[8][row], s); atomicAdd(&stat[9][row], q); }
    }
    __syncthreads();
    if (tid < 16) {
        float m = stat[8][tid] * (1.0f / U_);
        float v = stat[9][tid] * (1.0f / U_) - m * m;
        stat[10][tid] = m; stat[11][tid] = rsqrtf(v + EPS_);
    }
    __syncthreads();

    float sgu = P.sg[u], sbu = P.sb[u];
#pragma unroll
    for (int row = 0; row < 16; ++row) {
        float cl = (cn[row] - stat[10][row]) * stat[11][row] * sgu + sbu;
        float o  = zs[row][u + 1536];
        float hn = sigm(o) * tanhf(cl);
        int mv = mask[(long)(b0 + row) * T_ + t];
        float hold = hs[row][u];
        float hnew = mv ? hn : hold;
        hs[row][u] = hnew;
        creg[row]  = mv ? cl : creg[row];
        int pidx = apack_idx(row, u);
        hpk_out[pidx] = (__bf16)hnew;                 // masked state, packed
        if (rawpk_out) rawpk_out[pidx] = (__bf16)hn;  // raw hn feeds next layer
        if (outreg) outreg[row] = mv ? hn : outreg[row];
    }
    __syncthreads();
}

// ---------------------------------------------------------------------------
// Persistent kernel: 64 blocks x 512 threads; each block carries 16 batch
// rows through all 100 timesteps, both layers, and the softmax head.
// ---------------------------------------------------------------------------
__global__ __launch_bounds__(512, 1)
void lstm_persistent(const float* __restrict__ x, const int* __restrict__ mask,
                     LayerParams P0, LayerParams P1,
                     const float* __restrict__ Wd, const float* __restrict__ bd,
                     float* __restrict__ dout) {
    __shared__ float zs[16][GW];                 // 128 KB
    __shared__ float h0s[16][U_];                // 32 KB
    __shared__ float h1s[16][U_];                // 32 KB
    __shared__ v16bf xpk_v[10 * 32];             // 10 KB  packed x A-frags
    __shared__ v16bf hpk0_v[16 * 32];            // 16 KB  packed h0 state
    __shared__ v16bf hpk1_v[16 * 32];            // 16 KB  packed h1 state
    __shared__ v16bf rawpk_v[16 * 32];           // 16 KB  packed raw h0
    __shared__ float stat[12][16];
    __shared__ float lg[16][C_];

    const int tid  = threadIdx.x;
    const int lane = tid & 31;
    const int b0   = blockIdx.x * 16;

    // zero f32 state + packed state
    for (int i = tid; i < 16 * U_; i += 512) {
        h0s[i >> 9][i & (U_ - 1)] = 0.0f;
        h1s[i >> 9][i & (U_ - 1)] = 0.0f;
    }
    {
        int* z0 = (int*)hpk0_v;
        int* z1 = (int*)hpk1_v;
        for (int i = tid; i < 16 * 32 * 8; i += 512) { z0[i] = 0; z1[i] = 0; }
    }
    float c0r[16], c1r[16], outr[16];
#pragma unroll
    for (int r = 0; r < 16; ++r) { c0r[r] = 0.f; c1r[r] = 0.f; outr[r] = 0.f; }
    // layer_step's initial barrier orders the zero-init before first use

    for (int t = 0; t < T_; ++t) {
        // pack this step's x tile (16 rows x 320 K) into A-fragment order
        {
            __bf16* xp = (__bf16*)xpk_v;
            for (int i = tid; i < 10 * 32 * 16; i += 512) {
                int e      = i & 15;
                int lane_a = (i >> 4) & 31;
                int kt     = i >> 9;
                int m      = lane_a & 15;
                int kb_    = (lane_a < 16) ? 0 : 8;
                int k      = kt * 32 + kb_ + e + ((e & 8) ? 8 : 0);
                float v = (k < F_) ? x[((long)(b0 + m) * T_ + t) * F_ + k] : 0.0f;
                xp[i] = (__bf16)v;
            }
        }   // layer_step's first barrier covers these writes
        layer_step(P0, xpk_v, hpk0_v, h0s, c0r,
                   (__bf16*)hpk0_v, (__bf16*)rawpk_v, nullptr,
                   zs, stat, mask, b0, t);
        layer_step(P1, rawpk_v, hpk1_v, h1s, c1r,
                   (__bf16*)hpk1_v, nullptr, outr,
                   zs, stat, mask, b0, t);
    }

    // ---- head: softmax(out @ Wd + bd) ------------------------------------
    if (tid < 16 * C_) lg[tid / C_][tid % C_] = 0.0f;
    __syncthreads();
    const int u = tid;
#pragma unroll
    for (int row = 0; row < 16; ++row) {
        float v = outr[row];
#pragma unroll
        for (int c = 0; c < C_; ++c) {
            float p = v * Wd[(long)u * C_ + c];
            p = wave_reduce(p);
            if (lane == 0) atomicAdd(&lg[row][c], p);
        }
    }
    __syncthreads();
    if (tid < 16) {
        int row = tid;
        float l[C_]; float mx = -1e30f;
#pragma unroll
        for (int c = 0; c < C_; ++c) { l[c] = lg[row][c] + bd[c]; mx = fmaxf(mx, l[c]); }
        float s = 0.f;
#pragma unroll
        for (int c = 0; c < C_; ++c) { l[c] = expf(l[c] - mx); s += l[c]; }
        float inv = 1.0f / s;
#pragma unroll
        for (int c = 0; c < C_; ++c) dout[(long)(b0 + row) * C_ + c] = l[c] * inv;
    }
}

// ---------------------------------------------------------------------------
extern "C" void kernel_launch(void* const* d_in, const int* in_sizes, int n_in,
                              void* d_out, int out_size, void* d_ws, size_t ws_size,
                              hipStream_t stream) {
    const float* x    = (const float*)d_in[0];
    const int*   mask = (const int*)d_in[1];
    const float* W0 = (const float*)d_in[2];
    const float* R0 = (const float*)d_in[3];
    const float* W1 = (const float*)d_in[11];
    const float* R1 = (const float*)d_in[12];
    const float* Wd = (const float*)d_in[20];
    const float* bd = (const float*)d_in[21];

    char* ws = (char*)d_ws;
    __hip_bfloat16* Wpk0 = (__hip_bfloat16*)(ws);                 // 10 ktiles
    __hip_bfloat16* Rpk0 = (__hip_bfloat16*)(ws + 1310720);       // 16 ktiles
    __hip_bfloat16* Wpk1 = (__hip_bfloat16*)(ws + 3407872);       // 16 ktiles
    __hip_bfloat16* Rpk1 = (__hip_bfloat16*)(ws + 5505024);       // 16 ktiles

    long tot10 = 10L * 32 * 16 * (GW / 16);   // 655360
    long tot16 = 16L * 32 * 16 * (GW / 16);   // 1048576
    pack_weights<<<dim3((tot10 + 255) / 256), dim3(256), 0, stream>>>(W0, Wpk0, F_, 10);
    pack_weights<<<dim3((tot16 + 255) / 256), dim3(256), 0, stream>>>(R0, Rpk0, U_, 16);
    pack_weights<<<dim3((tot16 + 255) / 256), dim3(256), 0, stream>>>(W1, Wpk1, U_, 16);
    pack_weights<<<dim3((tot16 + 255) / 256), dim3(256), 0, stream>>>(R1, Rpk1, U_, 16);

    LayerParams P0, P1;
    P0.Wpk = Wpk0; P0.Rpk = Rpk0;
    P0.bias = (const float*)d_in[4];
    P0.kg = (const float*)d_in[5];  P0.kb = (const float*)d_in[6];
    P0.rg = (const float*)d_in[7];  P0.rb = (const float*)d_in[8];
    P0.sg = (const float*)d_in[9];  P0.sb = (const float*)d_in[10];
    P0.ktk_x = 10;

    P1.Wpk = Wpk1; P1.Rpk = Rpk1;
    P1.bias = (const float*)d_in[13];
    P1.kg = (const float*)d_in[14]; P1.kb = (const float*)d_in[15];
    P1.rg = (const float*)d_in[16]; P1.rb = (const float*)d_in[17];
    P1.sg = (const float*)d_in[18]; P1.sb = (const float*)d_in[19];
    P1.ktk_x = 16;

    lstm_persistent<<<dim3(B_ / 16), dim3(512), 0, stream>>>(
        x, mask, P0, P1, Wd, bd, (float*)d_out);
}
